// Attention_10582799417937
// MI455X (gfx1250) — compile-verified, matrix operand order checked by reading, not run on hardware
//
#include <hip/hip_runtime.h>
#include <cstdint>
#include <cstddef>

#define DEVFN __device__ __forceinline__

typedef __bf16 bf16;
typedef __bf16 bf16x16 __attribute__((ext_vector_type(16)));
typedef float  floatx8 __attribute__((ext_vector_type(8)));

// Problem constants (from reference)
constexpr int Bn = 32, Cc = 384, Hh = 32, Ww = 32;
constexpr int HEADS = 6;
constexpr int N1 = Bn * Hh * Ww;      // 32768 q positions
constexpr int N2 = Bn * 16 * 16;      // 8192 kv positions
constexpr float SCALE = 0.125f;       // 64^-0.5
constexpr float BN_EPS = 1e-5f;

DEVFN bf16 f2bf(float f) {            // round-to-nearest-even f32 -> bf16
  uint32_t u = __float_as_uint(f);
  uint32_t r = u + 0x7FFFu + ((u >> 16) & 1u);
  uint16_t hbits = (uint16_t)(r >> 16);
  bf16 out;
  __builtin_memcpy(&out, &hbits, 2);
  return out;
}

// Load a 16x32 bf16 WMMA fragment (A: lane=row M, or B: lane=col N; K contiguous).
// tile points at element [0][0]; ld is the row stride in elements (multiple of 8).
DEVFN bf16x16 load_frag(const bf16* tile, int ld) {
  int lane = threadIdx.x & 31;
  int row  = lane & 15;
  int koff = (lane & 16) ? 8 : 0;     // lanes 0-15: K {0..7,16..23}; 16-31: {8..15,24..31}
  const bf16* p = tile + (size_t)row * ld + koff;
  union { bf16x16 v; uint4 q[2]; } u;
  u.q[0] = *reinterpret_cast<const uint4*>(p);
  u.q[1] = *reinterpret_cast<const uint4*>(p + 16);
  return u.v;
}

DEVFN floatx8 wmma_bf16(bf16x16 a, bf16x16 b, floatx8 c) {
  return __builtin_amdgcn_wmma_f32_16x16x32_bf16(false, a, false, b, (short)0, c, false, false);
}

// ---------------- depthwise 3x3 + folded BN -> bf16 NHWC ----------------
__global__ void dwbn_kernel(const float* __restrict__ x, const float* __restrict__ wdw,
                            const float* __restrict__ gamma, const float* __restrict__ beta,
                            const float* __restrict__ mean, const float* __restrict__ var,
                            bf16* __restrict__ y, int stride, int Hout, int Wout) {
  int idx = blockIdx.x * blockDim.x + threadIdx.x;
  int total = Bn * Hout * Wout * Cc;
  if (idx >= total) return;
  int c = idx % Cc;
  int n = idx / Cc;
  int wo = n % Wout;
  int t  = n / Wout;
  int ho = t % Hout;
  int b  = t / Hout;
  const float* xp = x + ((size_t)b * Cc + c) * (Hh * Ww);
  const float* wp = wdw + c * 9;
  int hi0 = ho * stride - 1, wi0 = wo * stride - 1;
  float acc = 0.f;
#pragma unroll
  for (int kh = 0; kh < 3; ++kh) {
    int hi = hi0 + kh;
    if ((unsigned)hi >= (unsigned)Hh) continue;
#pragma unroll
    for (int kw = 0; kw < 3; ++kw) {
      int wi = wi0 + kw;
      if ((unsigned)wi >= (unsigned)Ww) continue;
      acc += xp[hi * Ww + wi] * wp[kh * 3 + kw];
    }
  }
  float inv = gamma[c] * rsqrtf(var[c] + BN_EPS);
  y[idx] = f2bf(acc * inv + (beta[c] - mean[c] * inv));
}

__global__ void f32_to_bf16_kernel(const float* __restrict__ in, bf16* __restrict__ out, int n) {
  int i = blockIdx.x * blockDim.x + threadIdx.x;
  if (i < n) out[i] = f2bf(in[i]);
}

// ---------------- WMMA GEMM: out[m][n] = sum_c W[m][c] * act[n][c] ----------------
// 256 threads = 8 waves; block tile 128(M) x 64(N); wave tile 32x32; K = 384.
// mode 0: write bf16 n-major out[n*M + m]; mode 1: write f32 NCHW + bias.
__global__ void gemm_wmma_kernel(const bf16* __restrict__ Wm, const bf16* __restrict__ act,
                                 int M, int mode, bf16* __restrict__ outb,
                                 float* __restrict__ outf, const float* __restrict__ bias,
                                 int spatial) {
  constexpr int K = 384;
  int lane = threadIdx.x & 31;
  int wv = threadIdx.x >> 5;
  int wm = wv >> 1, wn = wv & 1;
  int m_base = blockIdx.y * 128 + wm * 32;
  int n_base = blockIdx.x * 64 + wn * 32;
  const bf16* A0 = Wm + (size_t)m_base * K;
  const bf16* A1 = Wm + (size_t)(m_base + 16) * K;
  const bf16* Bt0 = act + (size_t)n_base * K;
  const bf16* Bt1 = act + (size_t)(n_base + 16) * K;
  floatx8 z = {0.f, 0.f, 0.f, 0.f, 0.f, 0.f, 0.f, 0.f};
  floatx8 acc00 = z, acc01 = z, acc10 = z, acc11 = z;
  for (int k0 = 0; k0 < K; k0 += 32) {
    if (k0 + 32 < K) {
      __builtin_prefetch(Bt0 + (size_t)k0 + 32, 0, 1);
      __builtin_prefetch(Bt1 + (size_t)k0 + 32, 0, 1);
    }
    bf16x16 a0 = load_frag(A0 + k0, K);
    bf16x16 a1 = load_frag(A1 + k0, K);
    bf16x16 b0 = load_frag(Bt0 + k0, K);
    bf16x16 b1 = load_frag(Bt1 + k0, K);
    acc00 = wmma_bf16(a0, b0, acc00);
    acc01 = wmma_bf16(a0, b1, acc01);
    acc10 = wmma_bf16(a1, b0, acc10);
    acc11 = wmma_bf16(a1, b1, acc11);
  }
  int hlf = lane >> 4, c0 = lane & 15;
  floatx8 accs[2][2] = {{acc00, acc01}, {acc10, acc11}};
#pragma unroll
  for (int mt = 0; mt < 2; ++mt)
#pragma unroll
    for (int nt = 0; nt < 2; ++nt) {
#pragma unroll
      for (int r = 0; r < 8; ++r) {
        int m = m_base + mt * 16 + r + 8 * hlf;
        int n = n_base + nt * 16 + c0;
        float v = accs[mt][nt][r];
        if (mode == 0) {
          outb[(size_t)n * M + m] = f2bf(v);
        } else {
          int bb = n / spatial, hw = n % spatial;
          outf[((size_t)bb * M + m) * spatial + hw] = v + bias[m];
        }
      }
    }
}

// ---------------- attention: per (b,h) Q[1024,64] x K[256,64] -> softmax -> x V ----------------
// grid.x = B*HEADS, grid.y = 16 (Q row blocks of 64); 128 threads = 4 waves, 16 Q rows each.
__global__ void attn_kernel(const bf16* __restrict__ q, const bf16* __restrict__ kv,
                            bf16* __restrict__ o) {
  __shared__ bf16 vt[64][264];              // V^T: [d][j], padded, 16B-aligned rows
  // vv (async V staging) is dead after the transpose barrier; alias it with pP.
  __shared__ union ShU {
    bf16 vv[256][64];                       // V rows as copied from global (32 KB)
    bf16 pP[4][16][264];                    // per-wave P tile, A-fragment source (33 KB)
  } shu;
  int bh = blockIdx.x;
  int qb = blockIdx.y;
  int b = bh / HEADS, h = bh % HEADS;
  int tid = threadIdx.x;
  int lane = tid & 31, wv = tid >> 5;

  // ---- phase 1: bulk-copy V rows into LDS with async global->LDS b128 (ASYNCcnt path)
  const bf16* vbase = kv + (size_t)(b * 256) * 768 + 384 + h * 64;  // 16B aligned
  for (int e = tid; e < 2048; e += 128) {   // 2048 chunks of 8 bf16 (16B)
    int j = e >> 3, dc = e & 7;
    uint32_t lds_off = (uint32_t)(uintptr_t)&shu.vv[j][dc * 8];
    uint64_t gaddr   = (uint64_t)(uintptr_t)(vbase + (size_t)j * 768 + dc * 8);
    asm volatile("global_load_async_to_lds_b128 %0, %1, off"
                 :: "v"(lds_off), "v"(gaddr) : "memory");
  }
  asm volatile("s_wait_asynccnt 0x0" ::: "memory");
  __syncthreads();

  // ---- phase 2: in-LDS transpose vv[j][d] -> vt[d][j]
  for (int e = tid; e < 2048; e += 128) {
    int j = e >> 3, dc = e & 7;
    union { uint4 qd; bf16 bb[8]; } u;
    u.qd = *reinterpret_cast<const uint4*>(&shu.vv[j][dc * 8]);
#pragma unroll
    for (int i = 0; i < 8; ++i) vt[dc * 8 + i][j] = u.bb[i];
  }
  __syncthreads();                          // vv dead from here on; shu.pP live below

  int i_base = qb * 64 + wv * 16;
  const bf16* qtile = q + (size_t)(b * 1024 + i_base) * 384 + h * 64;
  const bf16* kbase = kv + (size_t)(b * 256) * 768 + h * 64;

  floatx8 z = {0.f, 0.f, 0.f, 0.f, 0.f, 0.f, 0.f, 0.f};
  floatx8 S[16];
#pragma unroll
  for (int t = 0; t < 16; ++t) S[t] = z;

  // S = Q K^T  (K-dim = 64 -> two 32-wide steps; A fragments reused over all 16 j-tiles)
  bf16x16 a0 = load_frag(qtile, 384);
  bf16x16 a1 = load_frag(qtile + 32, 384);
#pragma unroll
  for (int t = 0; t < 16; ++t) {
    const bf16* kt = kbase + (size_t)(t * 16) * 768;
    bf16x16 b0 = load_frag(kt, 768);
    bf16x16 b1 = load_frag(kt + 32, 768);
    S[t] = wmma_bf16(a0, b0, S[t]);
    S[t] = wmma_bf16(a1, b1, S[t]);
  }

  // softmax over 256 cols; each lane holds 8 rows x 16 cols of its wave's 16x256 strip
  float mx[8], sm[8];
#pragma unroll
  for (int r = 0; r < 8; ++r) mx[r] = -3.4e38f;
#pragma unroll
  for (int t = 0; t < 16; ++t)
#pragma unroll
    for (int r = 0; r < 8; ++r) mx[r] = fmaxf(mx[r], S[t][r]);
#pragma unroll
  for (int msk = 1; msk < 16; msk <<= 1)
#pragma unroll
    for (int r = 0; r < 8; ++r) mx[r] = fmaxf(mx[r], __shfl_xor(mx[r], msk, 32));
#pragma unroll
  for (int r = 0; r < 8; ++r) sm[r] = 0.f;
#pragma unroll
  for (int t = 0; t < 16; ++t)
#pragma unroll
    for (int r = 0; r < 8; ++r) {
      float e = __expf((S[t][r] - mx[r]) * SCALE);
      S[t][r] = e;
      sm[r] += e;
    }
#pragma unroll
  for (int msk = 1; msk < 16; msk <<= 1)
#pragma unroll
    for (int r = 0; r < 8; ++r) sm[r] += __shfl_xor(sm[r], msk, 32);
  float inv[8];
#pragma unroll
  for (int r = 0; r < 8; ++r) inv[r] = 1.f / sm[r];

  // write normalized P to LDS in row-major [row][col] (C-layout -> A-layout conversion)
  int hlf = lane >> 4, c0 = lane & 15;
#pragma unroll
  for (int t = 0; t < 16; ++t)
#pragma unroll
    for (int r = 0; r < 8; ++r)
      shu.pP[wv][r + 8 * hlf][c0 + 16 * t] = f2bf(S[t][r] * inv[r]);
  __syncthreads();

  // O = P @ V  (K-dim = 256 -> eight 32-wide steps; B fragments from V^T in LDS)
  floatx8 O[4];
#pragma unroll
  for (int nt = 0; nt < 4; ++nt) O[nt] = z;
#pragma unroll
  for (int ks = 0; ks < 8; ++ks) {
    bf16x16 pa = load_frag(&shu.pP[wv][0][ks * 32], 264);
#pragma unroll
    for (int nt = 0; nt < 4; ++nt) {
      bf16x16 vb = load_frag(&vt[nt * 16][ks * 32], 264);
      O[nt] = wmma_bf16(pa, vb, O[nt]);
    }
  }

  // store O -> bf16 n-major [N1][384] at column h*64+d
#pragma unroll
  for (int nt = 0; nt < 4; ++nt)
#pragma unroll
    for (int r = 0; r < 8; ++r)
      o[(size_t)(b * 1024 + i_base + r + 8 * hlf) * 384 + h * 64 + nt * 16 + c0] = f2bf(O[nt][r]);
}

// ---------------- host launcher ----------------
extern "C" void kernel_launch(void* const* d_in, const int* in_sizes, int n_in,
                              void* d_out, int out_size, void* d_ws, size_t ws_size,
                              hipStream_t stream) {
  (void)in_sizes; (void)n_in; (void)out_size; (void)ws_size;
  const float* x       = (const float*)d_in[0];
  const float* q_dw    = (const float*)d_in[1];
  const float* q_gamma = (const float*)d_in[2];
  const float* q_beta  = (const float*)d_in[3];
  const float* q_mean  = (const float*)d_in[4];
  const float* q_var   = (const float*)d_in[5];
  const float* q_pw    = (const float*)d_in[6];
  const float* kv_dw   = (const float*)d_in[7];
  const float* kv_gamma= (const float*)d_in[8];
  const float* kv_beta = (const float*)d_in[9];
  const float* kv_mean = (const float*)d_in[10];
  const float* kv_var  = (const float*)d_in[11];
  const float* kv_pw   = (const float*)d_in[12];
  const float* out_w   = (const float*)d_in[13];
  const float* out_b   = (const float*)d_in[14];
  float* out = (float*)d_out;

  char* ws = (char*)d_ws;
  size_t off = 0;
  auto take = [&](size_t bytes) {
    char* p = ws + off;
    off += (bytes + 255) & ~(size_t)255;
    return p;
  };
  bf16* wq  = (bf16*)take((size_t)384 * 384 * 2);
  bf16* wkv = (bf16*)take((size_t)768 * 384 * 2);
  bf16* wo  = (bf16*)take((size_t)384 * 384 * 2);
  bf16* yq  = (bf16*)take((size_t)N1 * 384 * 2);
  bf16* ykv = (bf16*)take((size_t)N2 * 384 * 2);
  bf16* qb  = (bf16*)take((size_t)N1 * 384 * 2);
  bf16* kvb = (bf16*)take((size_t)N2 * 768 * 2);
  bf16* ob  = (bf16*)take((size_t)N1 * 384 * 2);

  f32_to_bf16_kernel<<<(384 * 384 + 255) / 256, 256, 0, stream>>>(q_pw, wq, 384 * 384);
  f32_to_bf16_kernel<<<(768 * 384 + 255) / 256, 256, 0, stream>>>(kv_pw, wkv, 768 * 384);
  f32_to_bf16_kernel<<<(384 * 384 + 255) / 256, 256, 0, stream>>>(out_w, wo, 384 * 384);

  dwbn_kernel<<<(N1 * Cc) / 256, 256, 0, stream>>>(x, q_dw, q_gamma, q_beta, q_mean, q_var,
                                                   yq, 1, 32, 32);
  dwbn_kernel<<<(N2 * Cc) / 256, 256, 0, stream>>>(x, kv_dw, kv_gamma, kv_beta, kv_mean, kv_var,
                                                   ykv, 2, 16, 16);

  gemm_wmma_kernel<<<dim3(N1 / 64, 384 / 128), 256, 0, stream>>>(wq, yq, 384, 0, qb, nullptr,
                                                                 nullptr, 1024);
  gemm_wmma_kernel<<<dim3(N2 / 64, 768 / 128), 256, 0, stream>>>(wkv, ykv, 768, 0, kvb, nullptr,
                                                                 nullptr, 256);

  attn_kernel<<<dim3(Bn * HEADS, 16), 128, 0, stream>>>(qb, kvb, ob);

  gemm_wmma_kernel<<<dim3(N1 / 64, 384 / 128), 256, 0, stream>>>(wo, ob, 384, 1, nullptr, out,
                                                                 out_b, 1024);
}
